// ConditionalFlow_39900246180519
// MI455X (gfx1250) — matrix operand order repaired and checked by reading, not compiled, not run on hardware
//
#include <hip/hip_runtime.h>
#include <cmath>

// ---------------------------------------------------------------------------
// Conditional normalizing flow log-prob, fused megakernel for gfx1250.
// All 64-wide GEMMs run on v_wmma_f32_16x16x32_f16 (f16 in, f32 accumulate).
// One workgroup (256 thr = 8 wave32) processes 128 samples; activations stay
// in LDS across the whole 8-layer flow.
// Weights are pre-packed once per launch (f32 -> padded f16 panels) into d_ws;
// B fragments stream straight from L2 (global_load_b128, broadcast across all
// blocks). Fallback path stages weights through LDS if d_ws is too small.
// ---------------------------------------------------------------------------

#define TILE   128
#define NTHR   256
#define HIDF   64
#define DIN    8
#define NTRD   4
#define TBv    2.0f
#define MINBW  0.001f
#define MINBH  0.001f
#define MINDD  0.001f
#define LOG2PI 1.8378770664093453f

typedef __attribute__((ext_vector_type(16))) _Float16 v16h;
typedef __attribute__((ext_vector_type(8)))  _Float16 v8h;
typedef __attribute__((ext_vector_type(8)))  float    v8f;

struct ResNetP {
  const float *b_in, *b_out, *w_in, *w_out;
  const float *b0[2], *b1[2], *bc[2], *w0[2], *w1[2], *wc[2];
};
struct LuP    { const float *bias, *lower, *udiag, *upper; };
struct LayerP { ResNetP aff; LuP lu1, lu2; ResNetP spl; };
struct FlowP {
  const float* inputs;
  const float* context;
  ResNetP cond;
  LayerP  layers[8];
};

// --------------------- packed-weight workspace layout ----------------------
// Walk order (matches resnet_apply's cursor): w_in, {w0,w1,(wc)} x2, w_out.
constexpr int WSH_COND_SZ  = 64 * 32 + 4 * 64 * 64 + 32 * 64;            // 20480
constexpr int WSH_AFF_SZ   = 64 * 64 + 2 * (64 * 64 + 64 * 64 + 64 * 32) + 16 * 64; // 25600
constexpr int WSH_SPL_SZ   = 64 * 64 + 2 * (64 * 64 + 64 * 64 + 64 * 32) + 96 * 64; // 30720
constexpr int WSH_LAYER_SZ = WSH_AFF_SZ + WSH_SPL_SZ;                    // 56320
constexpr size_t WS_HALF_TOTAL = (size_t)WSH_COND_SZ + 8 * (size_t)WSH_LAYER_SZ; // 471040
constexpr size_t WS_BYTES_NEEDED = WS_HALF_TOTAL * 2;                    // ~0.9 MB

// ------------------------- LDS partition (dynamic) -------------------------
constexpr int OFF_TP  = 0;                         // float [TILE][96] spline/affine params
constexpr int SZ_TP   = TILE * 96 * 4;
constexpr int OFF_X   = OFF_TP + SZ_TP;            // float [TILE][8] flow state
constexpr int SZ_X    = TILE * DIN * 4;
constexpr int OFF_LAD = OFF_X + SZ_X;              // float [TILE] log|det|
constexpr int SZ_LAD  = TILE * 4;
constexpr int OFF_H   = OFF_LAD + SZ_LAD;          // f16 [TILE][64] resnet h
constexpr int SZ_H    = TILE * HIDF * 2;
constexpr int OFF_T   = OFF_H  + SZ_H;             // f16 [TILE][64] resnet t
constexpr int OFF_G   = OFF_T  + SZ_H;             // f16 [TILE][64] gate
constexpr int OFF_CAT = OFF_G  + SZ_H;             // f16 [TILE][64] concat input
constexpr int OFF_CTX = OFF_CAT + SZ_H;            // f16 [TILE][32] context features
constexpr int SZ_CTX  = TILE * 32 * 2;
constexpr int OFF_W   = OFF_CTX + SZ_CTX;          // f16 [96][64] staged weights (fallback)
constexpr int SZ_W    = 96 * HIDF * 2;
constexpr int SMEM_WS  = OFF_W;                    // workspace path: ~124.5 KB
constexpr int SMEM_LDS = OFF_W + SZ_W;             // fallback path: ~136.5 KB

// --------------------------- fragment builders -----------------------------
// A (16-bit, 16x32): lanes 0-15 -> M=lane, halves {K0..7, K16..23};
//                    lanes 16-31 -> M=lane-16, halves {K8..15, K24..31}.
__device__ __forceinline__ v16h load_a(const _Float16* base, int KP, int mbase,
                                       int kt, bool relu) {
  const int lane = threadIdx.x & 31;
  const int hi   = lane >> 4;
  const _Float16* r = base + (mbase + (lane & 15)) * KP + kt * 32 + hi * 8;
  union { v16h v; v8h h2[2]; _Float16 e[16]; } a;
  a.h2[0] = *(const v8h*)(r);
  a.h2[1] = *(const v8h*)(r + 16);
  if (relu) {
#pragma unroll
    for (int j = 0; j < 16; ++j)
      a.e[j] = (a.e[j] > (_Float16)0.f) ? a.e[j] : (_Float16)0.f;
  }
  return a.v;
}

// B (16-bit, 32x16): lane n%16 holds column n; lanes 0-15 K={0..7, 16..23},
// lanes 16-31 K={8..15, 24..31} (matches A fragment K order).
__device__ __forceinline__ v16h load_b(const _Float16* wbuf, int KP, int nbase,
                                       int kt) {
  const int lane = threadIdx.x & 31;
  const int hi   = lane >> 4;
  const _Float16* r = wbuf + (nbase + (lane & 15)) * KP + kt * 32 + hi * 8;
  union { v16h v; v8h h2[2]; } b;
  b.h2[0] = *(const v8h*)(r);
  b.h2[1] = *(const v8h*)(r + 16);
  return b.v;
}

enum { EP_F16 = 0, EP_SIG = 1, EP_F32 = 2 };

// out = epilogue(A @ W^T + bias). A: [TILE][KP] f16 LDS. W: [NP][KP] f16
// (either global packed panel or LDS staging buffer; provenance is exact
// after inlining, so loads lower to global_load_b128 / ds_load_b128).
// Each wave owns one 16-row M tile and sweeps all N tiles.
__device__ __forceinline__ void gemm_tile(const _Float16* A, int KP, bool reluA,
                                          const _Float16* W, const float* bias,
                                          int N, int NP,
                                          _Float16* out16, float* out32,
                                          int ostride, int ep) {
  const int wave  = threadIdx.x >> 5;
  const int lane  = threadIdx.x & 31;
  const int mbase = wave * 16;
  const int nk    = KP >> 5;
  v16h af[2];
  for (int kt = 0; kt < nk; ++kt) af[kt] = load_a(A, KP, mbase, kt, reluA);
  for (int nt = 0; nt < NP / 16; ++nt) {
    v8f acc = {0.f, 0.f, 0.f, 0.f, 0.f, 0.f, 0.f, 0.f};
    for (int kt = 0; kt < nk; ++kt) {
      v16h bf = load_b(W, KP, nt * 16, kt);
      acc = __builtin_amdgcn_wmma_f32_16x16x32_f16(false, af[kt], false, bf,
                                                   (short)0, acc, false, false);
    }
    const int   n  = nt * 16 + (lane & 15);
    const float bv = (n < N) ? bias[n] : 0.f;
#pragma unroll
    for (int i = 0; i < 8; ++i) {
      const int m = mbase + (lane >> 4) * 8 + i;   // C layout: M = i + 8*(lane>>4)
      float v = acc[i] + bv;
      if (ep == EP_SIG) v = 1.f / (1.f + __expf(-v));
      if (ep == EP_F32) { if (n < N) out32[m * ostride + n] = v; }
      else              { out16[m * ostride + n] = (_Float16)v; }
    }
  }
}

// Stage f32 global weights [N][K] -> f16 LDS [NP][KP], zero padded (fallback).
__device__ __forceinline__ void stage_w(const float* W, int N, int K, int NP,
                                        int KP, _Float16* sw) {
  for (int i = threadIdx.x; i < NP * KP; i += NTHR) {
    const int n = i / KP, k = i % KP;
    sw[i] = (_Float16)((n < N && k < K) ? W[n * K + k] : 0.f);
  }
}

// WS=true: weights pre-packed at wbase, no staging, barriers only around
// cross-wave elementwise (each wave reads/writes only its own 16 rows of
// activations). WS=false: stage each matrix through sW with barriers.
template <bool WS>
__device__ __forceinline__ void resnet_apply(
    const ResNetP& p, int in_f, int KPin, const _Float16* inbuf, bool hasCtx,
    const _Float16* ctxbuf, int out_f, int NPout, _Float16* out16,
    float* out32, int ostride, const _Float16* __restrict__ wbase,
    _Float16* sW, _Float16* sH, _Float16* sT, _Float16* sG) {
  int wo = 0;
  auto wp = [&](const float* g, int N, int K, int NP, int KP) -> const _Float16* {
    const _Float16* r;
    if constexpr (WS) {
      r = wbase + wo;
    } else {
      stage_w(g, N, K, NP, KP, sW);
      __syncthreads();
      r = sW;
    }
    wo += NP * KP;
    return r;
  };

  {
    const _Float16* w = wp(p.w_in, HIDF, in_f, HIDF, KPin);
    gemm_tile(inbuf, KPin, false, w, p.b_in, HIDF, HIDF, sH, nullptr, HIDF, EP_F16);
    if constexpr (!WS) __syncthreads();
  }
  for (int b = 0; b < 2; ++b) {
    {
      const _Float16* w = wp(p.w0[b], HIDF, HIDF, HIDF, HIDF);
      gemm_tile(sH, HIDF, true, w, p.b0[b], HIDF, HIDF, sT, nullptr, HIDF, EP_F16);
      if constexpr (!WS) __syncthreads();
    }
    {
      const _Float16* w = wp(p.w1[b], HIDF, HIDF, HIDF, HIDF);
      gemm_tile(sT, HIDF, true, w, p.b1[b], HIDF, HIDF, sT, nullptr, HIDF, EP_F16);
      if constexpr (!WS) __syncthreads();
    }
    if (hasCtx) {
      const _Float16* w = wp(p.wc[b], HIDF, 32, HIDF, 32);
      gemm_tile(ctxbuf, 32, false, w, p.bc[b], HIDF, HIDF, sG, nullptr, HIDF, EP_SIG);
    }
    __syncthreads();  // elementwise below strides across all rows (cross-wave)
    if (hasCtx) {
      for (int i = threadIdx.x; i < TILE * HIDF; i += NTHR)
        sH[i] = (_Float16)((float)sH[i] + (float)sT[i] * (float)sG[i]);
    } else {
      for (int i = threadIdx.x; i < TILE * HIDF; i += NTHR)
        sH[i] = (_Float16)((float)sH[i] + (float)sT[i]);
    }
    __syncthreads();
  }
  {
    const _Float16* w = wp(p.w_out, out_f, HIDF, NPout, HIDF);
    gemm_tile(sH, HIDF, false, w, p.b_out, out_f, NPout, out16, out32, ostride,
              out32 ? EP_F32 : EP_F16);
    __syncthreads();  // per-sample consumers read across waves
  }
}

// y = reverse?(x) @ U^T @ L^T + bias ; lad += sum(log diag). Thread-per-sample.
__device__ __forceinline__ void lu_apply(const LuP& p, float* sX, float* sLAD,
                                         bool rev) {
  const int s = threadIdx.x;
  if (s < TILE) {
    float x[8], t[8], y[8];
#pragma unroll
    for (int d = 0; d < 8; ++d) x[d] = sX[s * 8 + (rev ? 7 - d : d)];
    float logdet = 0.f;
#pragma unroll
    for (int j = 0; j < 8; ++j) {
      const float ud = p.udiag[j];
      const float sp = (ud > 20.f) ? ud : log1pf(__expf(ud));
      const float dg = sp + 0.001f;
      logdet += __logf(dg);
      float acc = dg * x[j];
      for (int k = j + 1; k < 8; ++k) acc += p.upper[j * 8 + k] * x[k];
      t[j] = acc;
    }
#pragma unroll
    for (int i = 0; i < 8; ++i) {
      float acc = t[i] + p.bias[i];
      for (int j = 0; j < i; ++j) acc += p.lower[i * 8 + j] * t[j];
      y[i] = acc;
    }
#pragma unroll
    for (int d = 0; d < 8; ++d) sX[s * 8 + d] = y[d];
    sLAD[s] += logdet;
  }
  __syncthreads();
}

// cat = [x_id(4) | ctx(32) | zeros(28)] as f16 rows.
__device__ __forceinline__ void build_cat(const float* sX, const _Float16* sCTX,
                                          _Float16* sCAT) {
  for (int i = threadIdx.x; i < TILE * HIDF; i += NTHR) {
    const int r = i / HIDF, c = i % HIDF;
    float v;
    if (c < NTRD)            v = sX[r * 8 + NTRD + c];
    else if (c < NTRD + 32)  v = (float)sCTX[r * 32 + (c - NTRD)];
    else                     v = 0.f;
    sCAT[i] = (_Float16)v;
  }
  __syncthreads();
}

__device__ __forceinline__ void affine_elem(const float* sTP, float* sX,
                                            float* sLAD) {
  const int s = threadIdx.x;
  if (s < TILE) {
    float lad = 0.f;
#pragma unroll
    for (int d = 0; d < NTRD; ++d) {
      const float shift = sTP[s * 96 + d];
      const float sc = 1.f / (1.f + __expf(-(sTP[s * 96 + NTRD + d] + 2.f))) + 0.001f;
      sX[s * 8 + d] = sX[s * 8 + d] * sc + shift;
      lad += __logf(sc);
    }
    sLAD[s] += lad;
  }
  __syncthreads();
}

// Rational-quadratic spline, thread-per-sample; bin selection via unrolled
// select chains (no dynamically-indexed private arrays -> no scratch).
__device__ __forceinline__ void spline_elem(const float* sTP, float* sX,
                                            float* sLAD) {
  const int s = threadIdx.x;
  if (s < TILE) {
    float lad = 0.f;
    const float ssc = 0.125f;  // 1/sqrt(HID)
    for (int d = 0; d < NTRD; ++d) {
      const float* pp = sTP + s * 96 + d * 23;
      const float xv = sX[s * 8 + d];
      const bool inside = (xv >= -TBv) && (xv <= TBv);
      const float xc = fminf(fmaxf(xv, -TBv), TBv);

      float ew[8], eh[8];
      float mxw = -1e30f, mxh = -1e30f;
#pragma unroll
      for (int k = 0; k < 8; ++k) {
        mxw = fmaxf(mxw, pp[k] * ssc);
        mxh = fmaxf(mxh, pp[8 + k] * ssc);
      }
      float sw_ = 0.f, sh_ = 0.f;
#pragma unroll
      for (int k = 0; k < 8; ++k) {
        ew[k] = __expf(pp[k] * ssc - mxw);       sw_ += ew[k];
        eh[k] = __expf(pp[8 + k] * ssc - mxh);   sh_ += eh[k];
      }
      const float invw = (1.f - MINBW * 8.f) / sw_;
      const float invh = (1.f - MINBH * 8.f) / sh_;

      int idx = 0;
      float cwL = -TBv, wsel = 0.f;
      {
        float c = 0.f, cwprev = -TBv;
#pragma unroll
        for (int k = 0; k < 8; ++k) {
          c += MINBW + ew[k] * invw;
          const float cwk1 = (k == 7) ? TBv : (2.f * TBv * c - TBv);
          if (xc >= cwprev) { idx = k; cwL = cwprev; wsel = cwk1 - cwprev; }
          cwprev = cwk1;
        }
      }
      float chL = -TBv, hsel = 0.f;
      {
        float c = 0.f, chprev = -TBv;
#pragma unroll
        for (int k = 0; k < 8; ++k) {
          c += MINBH + eh[k] * invh;
          const float chk1 = (k == 7) ? TBv : (2.f * TBv * c - TBv);
          if (k == idx) { chL = chprev; hsel = chk1 - chprev; }
          chprev = chk1;
        }
      }
      // derivative pads are exactly 1.0 (MIN_D + softplus(DERIV_PAD))
      float d0v = 1.f, d1v = 1.f;
#pragma unroll
      for (int k = 0; k < 7; ++k) {
        const float u = pp[16 + k];
        const float sp = (u > 20.f) ? u : log1pf(__expf(u));
        const float dk = MINDD + sp;
        if (idx == k + 1) d0v = dk;
        if (idx == k)     d1v = dk;
      }
      const float th    = (xc - cwL) / wsel;
      const float delta = hsel / wsel;
      const float t1m   = th * (1.f - th);
      const float den   = delta + (d0v + d1v - 2.f * delta) * t1m;
      const float y     = chL + hsel * (delta * th * th + d0v * t1m) / den;
      const float dnum  = delta * delta *
          (d1v * th * th + 2.f * delta * t1m + d0v * (1.f - th) * (1.f - th));
      const float ladd  = __logf(dnum) - 2.f * __logf(den);
      sX[s * 8 + d] = inside ? y : xv;
      lad += inside ? ladd : 0.f;
    }
    sLAD[s] += lad;
  }
  __syncthreads();
}

// --------------------- weight pre-pack (one-time, tiny) --------------------
// Each block converts one f32 matrix into a zero-padded f16 [NP][KP] panel.
__global__ void __launch_bounds__(NTHR)
pack_weights(FlowP P, _Float16* __restrict__ ws) {
  const int id = blockIdx.x;
  const float* src = nullptr;
  int N = 0, K = 0, NP = 0, KP = 0, off = 0;
  if (id < 6) {
    switch (id) {
      case 0: src = P.cond.w_in;  N = 64; K = 6;  NP = 64; KP = 32; off = 0;     break;
      case 1: src = P.cond.w0[0]; N = 64; K = 64; NP = 64; KP = 64; off = 2048;  break;
      case 2: src = P.cond.w1[0]; N = 64; K = 64; NP = 64; KP = 64; off = 6144;  break;
      case 3: src = P.cond.w0[1]; N = 64; K = 64; NP = 64; KP = 64; off = 10240; break;
      case 4: src = P.cond.w1[1]; N = 64; K = 64; NP = 64; KP = 64; off = 14336; break;
      default:src = P.cond.w_out; N = 32; K = 64; NP = 32; KP = 64; off = 18432; break;
    }
  } else {
    const int t = id - 6, L = t / 16, r = t % 16;
    const bool isAff = (r < 8);
    const int rr = isAff ? r : r - 8;
    const int base = WSH_COND_SZ + L * WSH_LAYER_SZ + (isAff ? 0 : WSH_AFF_SZ);
    const ResNetP& rp = isAff ? P.layers[L].aff : P.layers[L].spl;
    N = 64; K = 64; NP = 64; KP = 64;
    switch (rr) {
      case 0: src = rp.w_in;  K = 36;                         off = base;         break;
      case 1: src = rp.w0[0];                                 off = base + 4096;  break;
      case 2: src = rp.w1[0];                                 off = base + 8192;  break;
      case 3: src = rp.wc[0]; K = 32; KP = 32;                off = base + 12288; break;
      case 4: src = rp.w0[1];                                 off = base + 14336; break;
      case 5: src = rp.w1[1];                                 off = base + 18432; break;
      case 6: src = rp.wc[1]; K = 32; KP = 32;                off = base + 22528; break;
      default:
        src = rp.w_out;
        N = isAff ? 8 : 92; NP = isAff ? 16 : 96;
        off = base + 24576;
        break;
    }
  }
  _Float16* dst = ws + off;
  for (int i = threadIdx.x; i < NP * KP; i += NTHR) {
    const int n = i / KP, k = i % KP;
    dst[i] = (_Float16)((n < N && k < K) ? src[n * K + k] : 0.f);
  }
}

// ------------------------------- main kernel -------------------------------
template <bool WS>
__global__ void __launch_bounds__(NTHR, 1)
flow_kernel(FlowP P, const _Float16* __restrict__ wsw, float* __restrict__ out) {
  extern __shared__ unsigned char smem[];
  float*    sTP  = (float*)(smem + OFF_TP);
  float*    sX   = (float*)(smem + OFF_X);
  float*    sLAD = (float*)(smem + OFF_LAD);
  _Float16* sH   = (_Float16*)(smem + OFF_H);
  _Float16* sT   = (_Float16*)(smem + OFF_T);
  _Float16* sG   = (_Float16*)(smem + OFF_G);
  _Float16* sCAT = (_Float16*)(smem + OFF_CAT);
  _Float16* sCTX = (_Float16*)(smem + OFF_CTX);
  _Float16* sW   = (_Float16*)(smem + OFF_W);   // used only when !WS

  const int s0 = blockIdx.x * TILE;

  for (int i = threadIdx.x; i < TILE * DIN; i += NTHR)
    sX[i] = P.inputs[s0 * DIN + i];
  if (threadIdx.x < TILE) sLAD[threadIdx.x] = 0.f;
  // raw context rows (6 wide, padded to 32) into sCAT used with KP=32
  for (int i = threadIdx.x; i < TILE * 32; i += NTHR) {
    const int r = i / 32, c = i % 32;
    sCAT[i] = (_Float16)((c < 6) ? P.context[(s0 + r) * 6 + c] : 0.f);
  }
  __syncthreads();

  // cond_net: 6 -> 32
  resnet_apply<WS>(P.cond, 6, 32, sCAT, false, nullptr, 32, 32, sCTX, nullptr,
                   32, wsw, sW, sH, sT, sG);

  for (int L = 0; L < 8; ++L) {
    const LayerP& lp = P.layers[L];
    const _Float16* waff = wsw + WSH_COND_SZ + L * WSH_LAYER_SZ;
    const _Float16* wspl = waff + WSH_AFF_SZ;

    lu_apply(lp.lu1, sX, sLAD, true);                        // reverse + LU1
    build_cat(sX, sCTX, sCAT);
    resnet_apply<WS>(lp.aff, 36, 64, sCAT, true, sCTX, 8, 16,   // affine net
                     nullptr, sTP, 96, waff, sW, sH, sT, sG);
    affine_elem(sTP, sX, sLAD);
    lu_apply(lp.lu2, sX, sLAD, true);                        // reverse + LU2
    build_cat(sX, sCTX, sCAT);
    resnet_apply<WS>(lp.spl, 36, 64, sCAT, true, sCTX, 92, 96,  // spline net
                     nullptr, sTP, 96, wspl, sW, sH, sT, sG);
    spline_elem(sTP, sX, sLAD);
  }

  if (threadIdx.x < TILE) {
    const int s = threadIdx.x;
    float ss = 0.f;
#pragma unroll
    for (int d = 0; d < DIN; ++d) { const float v = sX[s * DIN + d]; ss += v * v; }
    out[s0 + s] = -0.5f * ss - 0.5f * (float)DIN * LOG2PI + sLAD[s];
  }
}

// ------------------------------ host launch --------------------------------
extern "C" void kernel_launch(void* const* d_in, const int* in_sizes, int n_in,
                              void* d_out, int out_size, void* d_ws,
                              size_t ws_size, hipStream_t stream) {
  (void)out_size; (void)n_in;
  FlowP P;
  P.inputs  = (const float*)d_in[0];
  P.context = (const float*)d_in[1];

  int idx = 2;  // params leaves in jax pytree (sorted dict key) order
  auto nxt = [&]() { return (const float*)d_in[idx++]; };
  auto readResnet = [&](ResNetP& r, bool ctx) {
    r.b_in = nxt(); r.b_out = nxt();
    for (int b = 0; b < 2; ++b) {           // blocks: b0,b1,(bc),w0,w1,(wc)
      r.b0[b] = nxt(); r.b1[b] = nxt();
      if (ctx) r.bc[b] = nxt(); else r.bc[b] = nullptr;
      r.w0[b] = nxt(); r.w1[b] = nxt();
      if (ctx) r.wc[b] = nxt(); else r.wc[b] = nullptr;
    }
    r.w_in = nxt(); r.w_out = nxt();
  };
  auto readLu = [&](LuP& l) {               // bias, lower, udiag, upper
    l.bias = nxt(); l.lower = nxt(); l.udiag = nxt(); l.upper = nxt();
  };

  readResnet(P.cond, false);                // cond_net (no ctx gates)
  for (int L = 0; L < 8; ++L) {             // layers: affine, lu1, lu2, spline
    readResnet(P.layers[L].aff, true);
    readLu(P.layers[L].lu1);
    readLu(P.layers[L].lu2);
    readResnet(P.layers[L].spl, true);
  }

  const int batch = in_sizes[0] / DIN;      // 262144
  const int grid  = batch / TILE;           // 2048

  if (ws_size >= WS_BYTES_NEEDED && d_ws != nullptr) {
    _Float16* ws = (_Float16*)d_ws;
    pack_weights<<<134, NTHR, 0, stream>>>(P, ws);
    hipFuncSetAttribute(reinterpret_cast<const void*>(&flow_kernel<true>),
                        hipFuncAttributeMaxDynamicSharedMemorySize, SMEM_WS);
    flow_kernel<true><<<grid, NTHR, SMEM_WS, stream>>>(P, (const _Float16*)ws,
                                                       (float*)d_out);
  } else {
    hipFuncSetAttribute(reinterpret_cast<const void*>(&flow_kernel<false>),
                        hipFuncAttributeMaxDynamicSharedMemorySize, SMEM_LDS);
    flow_kernel<false><<<grid, NTHR, SMEM_LDS, stream>>>(
        P, (const _Float16*)nullptr, (float*)d_out);
  }
}
